// ClinicalT5STMoE_86698209837684
// MI455X (gfx1250) — matrix operand order repaired and checked by reading, not compiled, not run on hardware
//
#include <hip/hip_runtime.h>
#include <math.h>

typedef __bf16 bf16;
typedef __attribute__((ext_vector_type(16))) __bf16 v16bf;
typedef __attribute__((ext_vector_type(8))) float v8f;
typedef __attribute__((ext_vector_type(4))) unsigned int u32x4;
typedef __attribute__((ext_vector_type(8))) int i32x8;
typedef __attribute__((ext_vector_type(4))) int i32x4;

#define HAS_TDM (__has_builtin(__builtin_amdgcn_tensor_load_to_lds) && \
                 __has_builtin(__builtin_amdgcn_s_wait_tensorcnt))

constexpr int kD  = 768;
constexpr int kH  = 12;
constexpr int kDK = 64;
constexpr int kDFF = 3072;
constexpr int kE  = 8;
constexpr int kHM = 2048;
constexpr int kV  = 32128;
constexpr int kB  = 2;
constexpr int kSe = 512;
constexpr int kSd = 128;
constexpr int kTe = kB * kSe;   // 1024 encoder tokens
constexpr int kTd = kB * kSd;   // 256 decoder tokens

// ---------------------------------------------------------------- helpers

__device__ inline bf16 tobf(float x) { return (bf16)x; }

__device__ inline float geluf(float x) {
  float x3 = x * x * x;
  return 0.5f * x * (1.0f + tanhf(0.7978845608028654f * (x + 0.044715f * x3)));
}

// A fragment: 16x32 bf16, row-major source (lda in floats), contiguous k order:
// lanes 0-15: row=lane, K = 0..15; lanes 16-31: row=lane-16, K = 16..31
__device__ inline v16bf load_a_frag(const float* __restrict__ A, int lda) {
  int lane = threadIdx.x & 31;
  int row  = lane & 15;
  int kb   = (lane < 16) ? 0 : 16;
  const float* p = A + (size_t)row * lda + kb;
  v16bf f;
#pragma unroll
  for (int j = 0; j < 8; ++j) f[j] = tobf(p[j]);
#pragma unroll
  for (int j = 0; j < 8; ++j) f[8 + j] = tobf(p[8 + j]);
  return f;
}

// B fragment: 32x16 bf16 from row-major B[K,N] (ldb in floats), no edge guard.
// lanes 0-15: col=lane, K=0..15 ; lanes 16-31: col=lane-16, K=16..31
__device__ inline v16bf load_b_frag_n_full(const float* __restrict__ B, int ldb) {
  int lane = threadIdx.x & 31;
  int col  = lane & 15;
  int kb   = (lane < 16) ? 0 : 16;
  v16bf f;
#pragma unroll
  for (int j = 0; j < 16; ++j) f[j] = tobf(B[(size_t)(kb + j) * ldb + col]);
  return f;
}

__device__ inline v16bf load_b_frag_t_full(const float* __restrict__ B, int ldb) {
  int lane = threadIdx.x & 31;
  int col  = lane & 15;
  int kb   = (lane < 16) ? 0 : 16;
  const float* p = B + (size_t)col * ldb + kb;
  v16bf f;
#pragma unroll
  for (int j = 0; j < 16; ++j) f[j] = tobf(p[j]);
  return f;
}

// LDS B fragment (tile stride fixed at 128 floats)
__device__ inline v16bf load_b_frag_lds(const float* Bl) {
  int lane = threadIdx.x & 31;
  int col  = lane & 15;
  int kb   = (lane < 16) ? 0 : 16;
  v16bf f;
#pragma unroll
  for (int j = 0; j < 16; ++j) f[j] = tobf(Bl[(kb + j) * 128 + col]);
  return f;
}

// Guarded variants (edge tiles only: router N=8)
__device__ inline v16bf load_b_frag_n(const float* __restrict__ B, int ldb, int nvalid) {
  int lane = threadIdx.x & 31;
  int col  = lane & 15;
  int kb   = (lane < 16) ? 0 : 16;
  v16bf f;
  if (col < nvalid) {
#pragma unroll
    for (int j = 0; j < 16; ++j) f[j] = tobf(B[(size_t)(kb + j) * ldb + col]);
  } else {
#pragma unroll
    for (int j = 0; j < 16; ++j) f[j] = tobf(0.0f);
  }
  return f;
}

__device__ inline v16bf load_b_frag_t(const float* __restrict__ B, int ldb, int nvalid) {
  int lane = threadIdx.x & 31;
  int col  = lane & 15;
  int kb   = (lane < 16) ? 0 : 16;
  v16bf f;
  if (col < nvalid) {
    const float* p = B + (size_t)col * ldb + kb;
#pragma unroll
    for (int j = 0; j < 16; ++j) f[j] = tobf(p[j]);
  } else {
#pragma unroll
    for (int j = 0; j < 16; ++j) f[j] = tobf(0.0f);
  }
  return f;
}

__device__ inline v8f wmma_bf16(v16bf a, v16bf b, v8f c) {
  return __builtin_amdgcn_wmma_f32_16x16x32_bf16(false, a, false, b, (short)0, c, false, false);
}

// C store per ISA layout: lanes 0-15 -> N=lane, M=i ; lanes 16-31 -> N=lane-16, M=8+i
__device__ inline void store_c(float* __restrict__ C, int ldc, v8f acc, int nvalid,
                               int act, const float* __restrict__ res, float alpha) {
  int lane = threadIdx.x & 31;
  int n    = lane & 15;
  int mb   = (lane < 16) ? 0 : 8;
  if (n >= nvalid) return;
#pragma unroll
  for (int i = 0; i < 8; ++i) {
    float x = acc[i] * alpha;
    if (act == 1)      x = fmaxf(x, 0.0f);
    else if (act == 2) x = geluf(x);
    if (res) x += res[(size_t)(mb + i) * ldc + n];
    C[(size_t)(mb + i) * ldc + n] = x;
  }
}

// ---------------------------------------------------------------- GEMM kernels

// Generic path. One wave computes a 32x64 C tile: 2 A-frags + 4 B-frags ->
// 8 WMMAs per k-step. grid=(ceil(N/64), M/32), block=32.
template <int TRANSB, bool FULL>
__global__ void gemm_wmma_kernel(const float* __restrict__ A, const float* __restrict__ B,
                                 float* __restrict__ C, int N, int K,
                                 int lda, int ldb, int ldc,
                                 int act, const float* __restrict__ res, float alpha) {
  int n0 = blockIdx.x * 64;
  int m0 = blockIdx.y * 32;
  int lane = threadIdx.x & 31;
  const float* Ab = A + (size_t)m0 * lda;
  v8f acc[2][4] = {};
  for (int k0 = 0; k0 < K; k0 += 32) {
    v16bf a0 = load_a_frag(Ab + k0, lda);
    v16bf a1 = load_a_frag(Ab + (size_t)16 * lda + k0, lda);
    if (k0 + 32 < K) {   // prefetch next K-tile of A and B (global_prefetch_b8)
      __builtin_prefetch(Ab + (size_t)(lane & 15) * lda + k0 + 32, 0, 1);
      __builtin_prefetch(Ab + (size_t)((lane & 15) + 16) * lda + k0 + 32, 0, 1);
      if (TRANSB) {
        __builtin_prefetch(B + (size_t)(n0 + lane) * ldb + k0 + 32, 0, 1);
        __builtin_prefetch(B + (size_t)(n0 + 32 + lane) * ldb + k0 + 32, 0, 1);
      } else {
        __builtin_prefetch(B + (size_t)(k0 + 32 + lane) * ldb + n0, 0, 1);
        __builtin_prefetch(B + (size_t)(k0 + 32 + lane) * ldb + n0 + 32, 0, 1);
      }
    }
#pragma unroll
    for (int s = 0; s < 4; ++s) {
      int nn = n0 + s * 16;
      if (!FULL && nn >= N) continue;          // uniform branch; EXEC stays full
      v16bf b;
      if (FULL) {
        b = TRANSB ? load_b_frag_t_full(B + (size_t)nn * ldb + k0, ldb)
                   : load_b_frag_n_full(B + (size_t)k0 * ldb + nn, ldb);
      } else {
        int nv = N - nn; if (nv > 16) nv = 16;
        b = TRANSB ? load_b_frag_t(B + (size_t)nn * ldb + k0, ldb, nv)
                   : load_b_frag_n(B + (size_t)k0 * ldb + nn, ldb, nv);
      }
      acc[0][s] = wmma_bf16(a0, b, acc[0][s]);
      acc[1][s] = wmma_bf16(a1, b, acc[1][s]);
    }
  }
  const float* rb = res ? res + (size_t)m0 * ldc + n0 : nullptr;
  float* Cb = C + (size_t)m0 * ldc + n0;
#pragma unroll
  for (int ms = 0; ms < 2; ++ms) {
#pragma unroll
    for (int s = 0; s < 4; ++s) {
      int nn = n0 + s * 16;
      if (!FULL && nn >= N) continue;
      int nv = FULL ? 16 : ((N - nn) < 16 ? (N - nn) : 16);
      store_c(Cb + (size_t)ms * 16 * ldc + s * 16, ldc, acc[ms][s], nv, act,
              rb ? rb + (size_t)ms * 16 * ldc + s * 16 : nullptr, alpha);
    }
  }
}

// TDM path: workgroup = 2 waves, C tile 32(M) x 128(N). B weight tiles
// (32x128 fp32 = 16KB) stream into double-buffered LDS via the Tensor Data
// Mover (TENSORcnt), overlapped with WMMA compute. B row-major [K,N] only,
// N%128==0, K%32==0. grid=(N/128, M/32), block=64.
#if HAS_TDM
__global__ __launch_bounds__(64) void gemm_wmma_tdm_kernel(
    const float* __restrict__ A, const float* __restrict__ B, float* __restrict__ C,
    int N, int K, int lda, int ldb, int ldc,
    int act, const float* __restrict__ res, float alpha) {
  __shared__ float ldsB[2][32 * 128];
  int n0 = blockIdx.x * 128;
  int m0 = blockIdx.y * 32;
  int wave = threadIdx.x >> 5;
  int lane = threadIdx.x & 31;
  const float* Ab = A + (size_t)m0 * lda;

  // Build D# and issue TENSOR_LOAD_TO_LDS for rows [k0,k0+32) x cols [n0,n0+128)
  auto issue_tdm = [&](int k0, int buf) {
    unsigned long long ga = (unsigned long long)(uintptr_t)(B + (size_t)k0 * ldb + n0);
    unsigned dim1 = (unsigned)(K - k0);
    u32x4 g0;
    g0[0] = 1u;                                          // count=1, user D#
    g0[1] = (unsigned)(buf * 32 * 128 * 4);              // lds_addr (bytes)
    g0[2] = (unsigned)(ga & 0xFFFFFFFFull);              // global_addr[31:0]
    g0[3] = (unsigned)((ga >> 32) & 0x1FFFFFFull) | (2u << 30);  // addr[56:32], type=2
    i32x8 g1;
    g1[0] = (int)(2u << 16);                             // data_size=2 (4 bytes)
    g1[1] = (int)(((unsigned)ldb & 0xFFFFu) << 16);      // tensor_dim0[15:0]
    g1[2] = (int)(((unsigned)ldb >> 16) | ((dim1 & 0xFFFFu) << 16)); // dim0 hi | dim1 lo
    g1[3] = (int)((dim1 >> 16) | (128u << 16));          // dim1 hi | tile_dim0=128
    g1[4] = 32;                                          // tile_dim1=32, tile_dim2=0
    g1[5] = (int)(unsigned)ldb;                          // tensor_dim0_stride[31:0]
    g1[6] = 0;                                           // stride0 hi, stride1 lo
    g1[7] = 0;
    i32x4 gz = {0, 0, 0, 0};
#if __clang_major__ >= 23
    i32x8 gz8 = {0, 0, 0, 0, 0, 0, 0, 0};
    __builtin_amdgcn_tensor_load_to_lds(g0, g1, gz, gz, gz8, 0);
#else
    __builtin_amdgcn_tensor_load_to_lds(g0, g1, gz, gz, 0);
#endif
  };

  v8f acc[2][4] = {};
  if (wave == 0) issue_tdm(0, 0);
  for (int k0 = 0; k0 < K; k0 += 32) {
    int cur = (k0 >> 5) & 1;
    if (wave == 0) __builtin_amdgcn_s_wait_tensorcnt(0);  // buffer `cur` landed
    __syncthreads();                                      // visible to both waves
    if (wave == 0 && k0 + 32 < K) issue_tdm(k0 + 32, cur ^ 1);
    v16bf a0 = load_a_frag(Ab + k0, lda);
    v16bf a1 = load_a_frag(Ab + (size_t)16 * lda + k0, lda);
    if (k0 + 32 < K) {
      __builtin_prefetch(Ab + (size_t)(lane & 15) * lda + k0 + 32, 0, 1);
      __builtin_prefetch(Ab + (size_t)((lane & 15) + 16) * lda + k0 + 32, 0, 1);
    }
    const float* Bl = &ldsB[cur][wave * 64];
#pragma unroll
    for (int s = 0; s < 4; ++s) {
      v16bf b = load_b_frag_lds(Bl + s * 16);
      acc[0][s] = wmma_bf16(a0, b, acc[0][s]);
      acc[1][s] = wmma_bf16(a1, b, acc[1][s]);
    }
    __syncthreads();                 // done reading `cur` before TDM reuses it
  }
  int nw = n0 + wave * 64;
  const float* rb = res ? res + (size_t)m0 * ldc + nw : nullptr;
  float* Cb = C + (size_t)m0 * ldc + nw;
#pragma unroll
  for (int ms = 0; ms < 2; ++ms) {
#pragma unroll
    for (int s = 0; s < 4; ++s) {
      store_c(Cb + (size_t)ms * 16 * ldc + s * 16, ldc, acc[ms][s], 16, act,
              rb ? rb + (size_t)ms * 16 * ldc + s * 16 : nullptr, alpha);
    }
  }
}
#else
// Fallback (no TDM builtin on this toolchain): same tiling, direct global B.
__global__ __launch_bounds__(64) void gemm_wmma_tdm_kernel(
    const float* __restrict__ A, const float* __restrict__ B, float* __restrict__ C,
    int N, int K, int lda, int ldb, int ldc,
    int act, const float* __restrict__ res, float alpha) {
  int n0 = blockIdx.x * 128 + (int)(threadIdx.x >> 5) * 64;
  int m0 = blockIdx.y * 32;
  const float* Ab = A + (size_t)m0 * lda;
  v8f acc[2][4] = {};
  for (int k0 = 0; k0 < K; k0 += 32) {
    v16bf a0 = load_a_frag(Ab + k0, lda);
    v16bf a1 = load_a_frag(Ab + (size_t)16 * lda + k0, lda);
#pragma unroll
    for (int s = 0; s < 4; ++s) {
      v16bf b = load_b_frag_n_full(B + (size_t)k0 * ldb + n0 + s * 16, ldb);
      acc[0][s] = wmma_bf16(a0, b, acc[0][s]);
      acc[1][s] = wmma_bf16(a1, b, acc[1][s]);
    }
  }
  const float* rb = res ? res + (size_t)m0 * ldc + n0 : nullptr;
  float* Cb = C + (size_t)m0 * ldc + n0;
#pragma unroll
  for (int ms = 0; ms < 2; ++ms)
#pragma unroll
    for (int s = 0; s < 4; ++s)
      store_c(Cb + (size_t)ms * 16 * ldc + s * 16, ldc, acc[ms][s], 16, act,
              rb ? rb + (size_t)ms * 16 * ldc + s * 16 : nullptr, alpha);
}
#endif

// scores[b,h,q,k] = Q.Kt + relbias[h,q,k] + (1-mask[b,k])*-1e9
// q,k stored [B*L, D], head h at cols [h*64,h*64+64). 16x64 tile/wave.
// grid=(Lk/64, Lq/16, B*H)
__global__ void qk_wmma_kernel(const float* __restrict__ q, const float* __restrict__ km,
                               float* __restrict__ scores, const float* __restrict__ bias,
                               const int* __restrict__ mask, int Lq, int Lk) {
  int n0 = blockIdx.x * 64;
  int m0 = blockIdx.y * 16;
  int z = blockIdx.z, b = z / kH, h = z % kH;
  const float* Ab = q  + ((size_t)(b * Lq + m0)) * kD + h * kDK;
  const float* Bb = km + ((size_t)(b * Lk + n0)) * kD + h * kDK;
  v8f acc[4] = {};
  for (int k0 = 0; k0 < kDK; k0 += 32) {
    v16bf a = load_a_frag(Ab + k0, kD);
#pragma unroll
    for (int s = 0; s < 4; ++s) {
      v16bf bf_ = load_b_frag_t_full(Bb + (size_t)(s * 16) * kD + k0, kD);
      acc[s] = wmma_bf16(a, bf_, acc[s]);
    }
  }
  int lane = threadIdx.x & 31;
  int n = lane & 15;
  int mb = (lane < 16) ? 0 : 8;
  float* Cp = scores + ((size_t)z * Lq + m0) * Lk + n0;
  const float* bp = bias ? bias + ((size_t)h * Lq + m0) * Lk + n0 : nullptr;
#pragma unroll
  for (int s = 0; s < 4; ++s) {
    float mterm = mask ? (1.0f - (float)mask[b * Lk + n0 + s * 16 + n]) * -1e9f : 0.0f;
#pragma unroll
    for (int i = 0; i < 8; ++i) {
      float x = acc[s][i] + mterm;
      if (bp) x += bp[(size_t)(mb + i) * Lk + s * 16 + n];
      Cp[(size_t)(mb + i) * Lk + s * 16 + n] = x;
    }
  }
}

// out[b,q,h,dk] = sum_k P[b,h,q,k] * V[b,k,h,dk].  16x64 tile/wave.
// grid=(Lq/16, B*H)
__global__ void av_wmma_kernel(const float* __restrict__ p, const float* __restrict__ v,
                               float* __restrict__ out, int Lq, int Lk) {
  int m0 = blockIdx.x * 16;
  int z = blockIdx.y, b = z / kH, h = z % kH;
  const float* Ab = p + ((size_t)z * Lq + m0) * Lk;
  const float* Bb = v + (size_t)b * Lk * kD + h * kDK;
  v8f acc[4] = {};
  for (int k0 = 0; k0 < Lk; k0 += 32) {
    v16bf a = load_a_frag(Ab + k0, Lk);
#pragma unroll
    for (int s = 0; s < 4; ++s) {
      v16bf bf_ = load_b_frag_n_full(Bb + (size_t)k0 * kD + s * 16, kD);
      acc[s] = wmma_bf16(a, bf_, acc[s]);
    }
  }
  float* Cp = out + ((size_t)(b * Lq + m0)) * kD + h * kDK;
#pragma unroll
  for (int s = 0; s < 4; ++s)
    store_c(Cp + s * 16, kD, acc[s], 16, 0, nullptr, 1.0f);
}

// ---------------------------------------------------------------- elementwise / reduce

__global__ void embed_kernel(const int* __restrict__ ids, const float* __restrict__ emb,
                             float* __restrict__ out) {
  int t = blockIdx.x;
  const float* src = emb + (size_t)ids[t] * kD;
  for (int d = threadIdx.x; d < kD; d += 256) out[(size_t)t * kD + d] = src[d];
}

__global__ void rms_kernel(const float* __restrict__ x, const float* __restrict__ w,
                           float* __restrict__ out) {
  int t = blockIdx.x;
  __shared__ float red[256];
  const float* xr = x + (size_t)t * kD;
  float s = 0.0f;
  for (int d = threadIdx.x; d < kD; d += 256) { float v = xr[d]; s += v * v; }
  red[threadIdx.x] = s;
  __syncthreads();
  for (int off = 128; off > 0; off >>= 1) {
    if (threadIdx.x < off) red[threadIdx.x] += red[threadIdx.x + off];
    __syncthreads();
  }
  float inv = rsqrtf(red[0] / (float)kD + 1e-6f);
  for (int d = threadIdx.x; d < kD; d += 256) out[(size_t)t * kD + d] = xr[d] * inv * w[d];
}

__global__ void softmax_kernel(float* __restrict__ s, int Lk) {
  int row = blockIdx.x;
  float* p = s + (size_t)row * Lk;
  __shared__ float red[256];
  float m = -1e30f;
  for (int j = threadIdx.x; j < Lk; j += 256) m = fmaxf(m, p[j]);
  red[threadIdx.x] = m;
  __syncthreads();
  for (int off = 128; off > 0; off >>= 1) {
    if (threadIdx.x < off) red[threadIdx.x] = fmaxf(red[threadIdx.x], red[threadIdx.x + off]);
    __syncthreads();
  }
  m = red[0];
  __syncthreads();
  float sum = 0.0f;
  for (int j = threadIdx.x; j < Lk; j += 256) { float e = expf(p[j] - m); p[j] = e; sum += e; }
  red[threadIdx.x] = sum;
  __syncthreads();
  for (int off = 128; off > 0; off >>= 1) {
    if (threadIdx.x < off) red[threadIdx.x] += red[threadIdx.x + off];
    __syncthreads();
  }
  float inv = 1.0f / red[0];
  for (int j = threadIdx.x; j < Lk; j += 256) p[j] *= inv;
}

__global__ void zero_kernel(float* __restrict__ p, unsigned n) {
  unsigned i = blockIdx.x * 256u + threadIdx.x;
  if (i < n) p[i] = 0.0f;
}

__global__ void add_kernel(const float* __restrict__ a, const float* __restrict__ b,
                           float* __restrict__ c, unsigned n) {
  unsigned i = blockIdx.x * 256u + threadIdx.x;
  if (i < n) c[i] = a[i] + b[i];
}

__global__ void geglu_kernel(const float* __restrict__ hg, float* __restrict__ act, unsigned n) {
  unsigned i = blockIdx.x * 256u + threadIdx.x;
  if (i >= n) return;
  unsigned t = i / kHM, j = i % kHM;
  float a = hg[(size_t)t * (2 * kHM) + j];
  float g = hg[(size_t)t * (2 * kHM) + kHM + j];
  act[i] = a * geluf(g);
}

__global__ void moe_acc_kernel(const float* __restrict__ eo, const float* __restrict__ comb,
                               int e, float* __restrict__ y, unsigned n) {
  unsigned i = blockIdx.x * 256u + threadIdx.x;
  if (i >= n) return;
  unsigned t = i / kD;
  y[i] += comb[(size_t)t * kE + e] * eo[i];
}

__global__ void router_kernel(const float* __restrict__ logits, float* __restrict__ comb,
                              float* __restrict__ macc, int T) {
  int t = blockIdx.x * 64 + threadIdx.x;
  if (t >= T) return;
  float l[kE], pr[kE];
  float mx = -1e30f;
  for (int e = 0; e < kE; ++e) { l[e] = logits[(size_t)t * kE + e]; mx = fmaxf(mx, l[e]); }
  float se = 0.0f;
  for (int e = 0; e < kE; ++e) { pr[e] = expf(l[e] - mx); se += pr[e]; }
  float lse = mx + logf(se);
  float inv = 1.0f / se;
  for (int e = 0; e < kE; ++e) pr[e] *= inv;
  int i0 = 0;
  for (int e = 1; e < kE; ++e) if (pr[e] > pr[i0]) i0 = e;
  int i1 = (i0 == 0) ? 1 : 0;
  for (int e = 0; e < kE; ++e) if (e != i0 && pr[e] > pr[i1]) i1 = e;
  float g0 = pr[i0];
  float g1 = pr[i1] > 0.2f ? pr[i1] : 0.0f;
  for (int e = 0; e < kE; ++e) comb[(size_t)t * kE + e] = 0.0f;
  comb[(size_t)t * kE + i0] = g0;
  comb[(size_t)t * kE + i1] = g1;
  atomicAdd(&macc[i0], 1.0f);
  for (int e = 0; e < kE; ++e) atomicAdd(&macc[kE + e], pr[e]);
  atomicAdd(&macc[16], lse * lse);
}

__global__ void aux_kernel(const float* __restrict__ macc, float* __restrict__ acc, float T) {
  if (threadIdx.x == 0 && blockIdx.x == 0) {
    float bal = 0.0f;
    for (int e = 0; e < kE; ++e) bal += (macc[e] / T) * (macc[kE + e] / T);
    bal *= (float)kE;
    acc[0] += 0.01f * bal + 0.001f * (macc[16] / T);
  }
}

__global__ void relbias_enc_kernel(const float* __restrict__ table, float* __restrict__ bias) {
  int idx = blockIdx.x * 256 + threadIdx.x;
  if (idx >= kSe * kSe) return;
  int i = idx / kSe, j = idx % kSe;
  int rel = j - i;
  int ret = rel > 0 ? 16 : 0;
  int n = rel < 0 ? -rel : rel;
  int bkt;
  if (n < 8) bkt = n;
  else {
    int lg = 8 + (int)(logf((float)n / 8.0f + 1e-6f) * (8.0f / logf(16.0f)));
    bkt = lg < 15 ? lg : 15;
  }
  bkt += ret;
  for (int h = 0; h < kH; ++h)
    bias[((size_t)h * kSe + i) * kSe + j] = table[bkt * kH + h];
}

__global__ void relbias_dec_kernel(const float* __restrict__ table, float* __restrict__ bias) {
  int idx = blockIdx.x * 256 + threadIdx.x;
  if (idx >= kSd * kSd) return;
  int i = idx / kSd, j = idx % kSd;
  int n = (i - j) > 0 ? (i - j) : 0;
  int bkt;
  if (n < 16) bkt = n;
  else {
    int lg = 16 + (int)(logf((float)n / 16.0f + 1e-6f) * (16.0f / logf(8.0f)));
    bkt = lg < 31 ? lg : 31;
  }
  float causal = (j > i) ? -1e9f : 0.0f;
  for (int h = 0; h < kH; ++h)
    bias[((size_t)h * kSd + i) * kSd + j] = table[bkt * kH + h] + causal;
}

__global__ void ce_kernel(const float* __restrict__ logits, const int* __restrict__ labels,
                          float* __restrict__ acc) {
  int t = blockIdx.x;
  const float* p = logits + (size_t)t * kV;
  __shared__ float red[256];
  float m = -1e30f;
  for (int j = threadIdx.x; j < kV; j += 256) m = fmaxf(m, p[j]);
  red[threadIdx.x] = m;
  __syncthreads();
  for (int off = 128; off > 0; off >>= 1) {
    if (threadIdx.x < off) red[threadIdx.x] = fmaxf(red[threadIdx.x], red[threadIdx.x + off]);
    __syncthreads();
  }
  m = red[0];
  __syncthreads();
  float s = 0.0f;
  for (int j = threadIdx.x; j < kV; j += 256) s += expf(p[j] - m);
  red[threadIdx.x] = s;
  __syncthreads();
  for (int off = 128; off > 0; off >>= 1) {
    if (threadIdx.x < off) red[threadIdx.x] += red[threadIdx.x + off];
    __syncthreads();
  }
  if (threadIdx.x == 0) {
    float lse = m + logf(red[0]);
    float ll = p[labels[t]];
    atomicAdd(acc, (lse - ll) / (float)kTd);
  }
}

__global__ void finalize_kernel(const float* __restrict__ acc, float* __restrict__ out) {
  if (threadIdx.x == 0 && blockIdx.x == 0) out[0] = acc[0];
}

// ---------------------------------------------------------------- host

extern "C" void kernel_launch(void* const* d_in, const int* in_sizes, int n_in,
                              void* d_out, int out_size, void* d_ws, size_t ws_size,
                              hipStream_t stream) {
  (void)in_sizes; (void)n_in; (void)out_size; (void)ws_size;

  const int* input_ids = (const int*)d_in[0];
  const int* attn_mask = (const int*)d_in[1];
  const int* dec_ids   = (const int*)d_in[2];
  const int* labels    = (const int*)d_in[3];

  // ---- params, JAX pytree (sorted dict keys) order ----
  int pidx = 4;
  auto Pp = [&]() { return (const float*)d_in[pidx++]; };
  struct AttnP { const float *ln, *wk, *wo, *wq, *wv; };
  struct MoeP  { const float *ff_wi, *ff_wo, *ln, *ln_ff, *ln_moe, *router, *w1, *w2; };
  struct FfnP  { const float *ln, *wi, *wo; };
  auto isMoe = [](int i) { return i == 0 || i == 2 || i == 4; };

  AttnP dattn[12], dcross[12], eattn[12];
  MoeP  dmoe[12], emoe[12];
  FfnP  dffn[12], effn[12];
  for (int i = 0; i < 12; ++i) {               // 'dec'
    dattn[i]  = AttnP{Pp(), Pp(), Pp(), Pp(), Pp()};     // ln, wk, wo, wq, wv
    dcross[i] = AttnP{Pp(), Pp(), Pp(), Pp(), Pp()};
    if (isMoe(i)) dmoe[i] = MoeP{Pp(), Pp(), Pp(), Pp(), Pp(), Pp(), Pp(), Pp()};
    else          dffn[i] = FfnP{Pp(), Pp(), Pp()};
  }
  const float* dec_ln_f = Pp();
  const float* dec_relb = Pp();
  const float* emb      = Pp();
  for (int i = 0; i < 12; ++i) {               // 'enc'
    eattn[i] = AttnP{Pp(), Pp(), Pp(), Pp(), Pp()};
    if (isMoe(i)) emoe[i] = MoeP{Pp(), Pp(), Pp(), Pp(), Pp(), Pp(), Pp(), Pp()};
    else          effn[i] = FfnP{Pp(), Pp(), Pp()};
  }
  const float* enc_ln_f = Pp();
  const float* enc_relb = Pp();

  // ---- workspace bump allocator ----
  float* wp = (float*)d_ws;
  auto alloc = [&](size_t n) { float* p = wp; wp += n; return p; };
  const size_t TD = (size_t)kTe * kD;
  float* xb = alloc(TD);   float* hb = alloc(TD);   float* h2 = alloc(TD);
  float* n1 = alloc(TD);   float* n2 = alloc(TD);
  float* qb = alloc(TD);   float* kb = alloc(TD);   float* vb = alloc(TD);
  float* ab = alloc(TD);   float* yb = alloc(TD);   float* mb = alloc(TD);
  float* m2 = alloc(TD);   float* t1 = alloc(TD);   float* enc_out = alloc(TD);
  float* relbE  = alloc((size_t)kH * kSe * kSe);
  float* relbD  = alloc((size_t)kH * kSd * kSd);
  float* scores = alloc((size_t)kB * kH * kSe * kSe);
  size_t bigN = (size_t)kTd * kV;                 // 8.22M floats: logits / hgate / ff
  if ((size_t)kTe * 2 * kHM > bigN) bigN = (size_t)kTe * 2 * kHM;
  float* big  = alloc(bigN);
  float* actb = alloc((size_t)kTe * kHM);
  float* rlog = alloc((size_t)kTe * kE);
  float* comb = alloc((size_t)kTe * kE);
  float* acc  = alloc(32);
  float* macc = alloc(32);

  // ---- launch helpers ----
  auto zero = [&](float* p, size_t n) {
    zero_kernel<<<dim3((unsigned)((n + 255) / 256)), dim3(256), 0, stream>>>(p, (unsigned)n);
  };
  auto add = [&](const float* a, const float* b, float* c, size_t n) {
    add_kernel<<<dim3((unsigned)((n + 255) / 256)), dim3(256), 0, stream>>>(a, b, c, (unsigned)n);
  };
  auto rms = [&](const float* x, const float* g, float* o, int T) {
    rms_kernel<<<dim3(T), dim3(256), 0, stream>>>(x, g, o);
  };
  auto gemm = [&](const float* A, const float* B, float* C, int M, int N, int K,
                  int lda, int ldb, int ldc, int transB, int act, const float* res, float alpha) {
    if (!transB && (N % 128) == 0) {   // TDM/LDS-staged path (weight streamers)
      gemm_wmma_tdm_kernel<<<dim3(N / 128, M / 32), dim3(64), 0, stream>>>(
          A, B, C, N, K, lda, ldb, ldc, act, res, alpha);
      return;
    }
    dim3 g((N + 63) / 64, M / 32);
    bool full = (N % 64) == 0;
    if (transB) {
      if (full) gemm_wmma_kernel<1, true ><<<g, dim3(32), 0, stream>>>(A, B, C, N, K, lda, ldb, ldc, act, res, alpha);
      else      gemm_wmma_kernel<1, false><<<g, dim3(32), 0, stream>>>(A, B, C, N, K, lda, ldb, ldc, act, res, alpha);
    } else {
      if (full) gemm_wmma_kernel<0, true ><<<g, dim3(32), 0, stream>>>(A, B, C, N, K, lda, ldb, ldc, act, res, alpha);
      else      gemm_wmma_kernel<0, false><<<g, dim3(32), 0, stream>>>(A, B, C, N, K, lda, ldb, ldc, act, res, alpha);
    }
  };

  auto attention = [&](const float* qin, const float* kvpre, const AttnP& p,
                       const float* biasT, const int* mask, int Lq, int Lk, float* outB) {
    rms(qin, p.ln, n1, kB * Lq);
    const float* kv = kvpre ? kvpre : n1;
    gemm(n1, p.wq, qb, kB * Lq, kD, kD, kD, kD, kD, 0, 0, nullptr, 1.0f);
    gemm(kv, p.wk, kb, kB * Lk, kD, kD, kD, kD, kD, 0, 0, nullptr, 1.0f);
    gemm(kv, p.wv, vb, kB * Lk, kD, kD, kD, kD, kD, 0, 0, nullptr, 1.0f);
    qk_wmma_kernel<<<dim3(Lk / 64, Lq / 16, kB * kH), dim3(32), 0, stream>>>(
        qb, kb, scores, biasT, mask, Lq, Lk);
    softmax_kernel<<<dim3(kB * kH * Lq), dim3(256), 0, stream>>>(scores, Lk);
    av_wmma_kernel<<<dim3(Lq / 16, kB * kH), dim3(32), 0, stream>>>(
        scores, vb, ab, Lq, Lk);
    gemm(ab, p.wo, outB, kB * Lq, kD, kD, kD, kD, kD, 0, 0, qin, 1.0f);
  };

  auto moe_block = [&](const float* hin, const MoeP& p, int T, float* xout) {
    rms(hin, p.ln, n1, T);                  // n
    rms(n1, p.ln_moe, n2, T);               // nm
    gemm(n2, p.router, rlog, T, kE, kD, kD, kE, kE, 0, 0, nullptr, 1.0f);
    zero(macc, 32);
    zero(yb, (size_t)T * kD);
    router_kernel<<<dim3((T + 63) / 64), dim3(64), 0, stream>>>(rlog, comb, macc, T);
    for (int e = 0; e < kE; ++e) {
      gemm(n2, p.w1 + (size_t)e * kD * 2 * kHM, big, T, 2 * kHM, kD,
           kD, 2 * kHM, 2 * kHM, 0, 0, nullptr, 1.0f);
      geglu_kernel<<<dim3((unsigned)(((size_t)T * kHM + 255) / 256)), dim3(256), 0, stream>>>(
          big, actb, (unsigned)((size_t)T * kHM));
      gemm(actb, p.w2 + (size_t)e * kHM * kD, t1, T, kD, kHM,
           kHM, kD, kD, 0, 0, nullptr, 1.0f);
      moe_acc_kernel<<<dim3((unsigned)(((size_t)T * kD + 255) / 256)), dim3(256), 0, stream>>>(
          t1, comb, e, yb, (unsigned)((size_t)T * kD));
    }
    aux_kernel<<<dim3(1), dim3(1), 0, stream>>>(macc, acc, (float)T);
    add(n1, yb, mb, (size_t)T * kD);        // m = n + y
    rms(mb, p.ln_ff, n2, T);
    gemm(n2, p.ff_wi, big, T, kDFF, kD, kD, kDFF, kDFF, 0, 2 /*gelu*/, nullptr, 1.0f);
    gemm(big, p.ff_wo, m2, T, kD, kDFF, kDFF, kD, kD, 0, 0, mb, 1.0f);  // m += ff
    add(hin, m2, xout, (size_t)T * kD);     // x = h + m
  };

  auto dense_ffn = [&](const float* hin, const FfnP& p, int T, float* xout) {
    rms(hin, p.ln, n1, T);
    gemm(n1, p.wi, big, T, kDFF, kD, kD, kDFF, kDFF, 0, 1 /*relu*/, nullptr, 1.0f);
    gemm(big, p.wo, xout, T, kD, kDFF, kDFF, kD, kD, 0, 0, hin, 1.0f);
  };

  // ---- forward pass ----
  zero(acc, 32);
  relbias_enc_kernel<<<dim3((kSe * kSe + 255) / 256), dim3(256), 0, stream>>>(enc_relb, relbE);
  relbias_dec_kernel<<<dim3((kSd * kSd + 255) / 256), dim3(256), 0, stream>>>(dec_relb, relbD);

  // encoder
  embed_kernel<<<dim3(kTe), dim3(256), 0, stream>>>(input_ids, emb, xb);
  for (int i = 0; i < 12; ++i) {
    attention(xb, nullptr, eattn[i], relbE, attn_mask, kSe, kSe, hb);
    if (isMoe(i)) moe_block(hb, emoe[i], kTe, xb);
    else          dense_ffn(hb, effn[i], kTe, xb);
  }
  rms(xb, enc_ln_f, enc_out, kTe);

  // decoder
  embed_kernel<<<dim3(kTd), dim3(256), 0, stream>>>(dec_ids, emb, xb);
  for (int i = 0; i < 12; ++i) {
    attention(xb, nullptr, dattn[i], relbD, nullptr, kSd, kSd, hb);
    attention(hb, enc_out, dcross[i], nullptr, attn_mask, kSd, kSe, h2);
    if (isMoe(i)) moe_block(h2, dmoe[i], kTd, xb);
    else          dense_ffn(h2, dffn[i], kTd, xb);
  }
  rms(xb, dec_ln_f, n1, kTd);

  // tied lm head with T5 rescale: logits = (dec_out * D^-0.5) @ emb^T
  gemm(n1, emb, big, kTd, kV, kD, kD, kD, kV, 1 /*transB*/, 0, nullptr,
       0.03608439182435161f);
  ce_kernel<<<dim3(kTd), dim3(256), 0, stream>>>(big, labels, acc);
  finalize_kernel<<<dim3(1), dim3(1), 0, stream>>>(acc, (float*)d_out);
}
